// LSTM_Model_10651518894170
// MI455X (gfx1250) — matrix operand order repaired and checked by reading, not compiled, not run on hardware
//
#include <hip/hip_runtime.h>
#include <math.h>

#define Hh 49      // hidden
#define Lh 7       // layers
#define Ih 7       // input size
#define Oh 7       // output size
#define Bh 1024
#define Th 512
#define HP 64      // padded hidden
#define GP 256     // padded gates = 4*HP
#define KTOT 128   // padded concat K: 64 (x part) + 64 (h part)
#define LDS_HSTRIDE 72    // halfs per row (16B aligned, bank-spread)
#define LDS_GSTRIDE 260   // floats per row

typedef __attribute__((ext_vector_type(16))) _Float16 v16h;
typedef __attribute__((ext_vector_type(8)))  _Float16 v8h;
typedef __attribute__((ext_vector_type(8)))  float    v8f;

// Native gfx1250 V_TANH_F32 when the builtin exists; otherwise a branchless
// exp-based expansion (no EXEC divergence, saturates correctly at +/-inf).
__device__ __forceinline__ float tanh_fast(float x) {
#if __has_builtin(__builtin_amdgcn_tanhf)
  return __builtin_amdgcn_tanhf(x);
#elif __has_builtin(__builtin_amdgcn_tanh_f32)
  return __builtin_amdgcn_tanh_f32(x);
#else
  return 1.0f - 2.0f / (1.0f + __expf(2.0f * x));
#endif
}

__device__ __forceinline__ float sigmoid_fast(float x) {
  return 0.5f * tanh_fast(0.5f * x) + 0.5f;
}

// Build one 16x32 f16 A-fragment from an LDS tile (row-major, LDS_HSTRIDE halfs/row).
// ISA A layout (16-bit, 16x32): lanes 0-15: M=lane, K={0..7,16..23};
// lanes 16-31: M=lane-16, K={8..15,24..31}. Two b128 LDS loads per lane.
__device__ __forceinline__ v16h load_a_frag(const _Float16* buf, int lane, int kb) {
  const int hs = lane >> 4;
  const int m  = lane & 15;
  const _Float16* p = buf + m * LDS_HSTRIDE + kb + hs * 8;
  v8h lo = *(const v8h*)p;          // K = kb + hs*8 .. +7
  v8h hi = *(const v8h*)(p + 16);   // K = kb + 16 + hs*8 .. +7
  v16h a;
  #pragma unroll
  for (int i = 0; i < 8; ++i) { a[i] = lo[i]; a[i + 8] = hi[i]; }
  return a;
}

// One LSTM layer: grid = B/16 workgroups x 128 threads (4 waves).
// Wave g computes gate g's 16x64 slice: 4 N-tiles x 4 K-tiles of wmma_f32_16x16x32_f16.
__global__ __launch_bounds__(128)
void lstm_layer_kernel(const _Float16* __restrict__ in,   // [B][T][HP] f16
                       _Float16* __restrict__ out,        // [B][T][HP] f16
                       const _Float16* __restrict__ W,    // [KTOT][GP] f16 (concat [x|h])
                       const float* __restrict__ bias) {  // [GP]
  __shared__ _Float16 xbuf[16 * LDS_HSTRIDE];
  __shared__ _Float16 hbuf[16 * LDS_HSTRIDE];
  __shared__ float    gbuf[16 * LDS_GSTRIDE];

  const int tid  = threadIdx.x;
  const int lane = tid & 31;
  const int w    = tid >> 5;               // wave id == gate id (i,f,g,o)
  const int b0   = blockIdx.x * 16;        // batch tile base

  // ---- weight (B-matrix) fragments: resident in VGPRs for the whole sequence.
  // B layout (16-bit, 32x16): lanes 0-15 hold N=lane, K={0..7,16..23};
  // lanes 16-31 hold N=lane-16, K={8..15,24..31} (matches A fragment order).
  v16h Bfrag[4][4];
  {
    const int nB    = lane & 15;
    const int halfB = lane >> 4;
    #pragma unroll
    for (int j = 0; j < 4; ++j) {
      #pragma unroll
      for (int kt = 0; kt < 4; ++kt) {
        const int ncol = (w * 4 + j) * 16 + nB;
        const int kb   = kt * 32 + halfB * 8;
        v16h b;
        #pragma unroll
        for (int e = 0; e < 16; ++e)
          b[e] = W[(size_t)(kb + (e & 7) + ((e >> 3) << 4)) * GP + ncol];
        Bfrag[j][kt] = b;
      }
    }
  }

  // ---- per-thread cell state: row m, cols n0..n0+7 of the 16x64 state tile
  const int m  = tid >> 3;
  const int n0 = (tid & 7) * 8;
  float cc[8];
  float bi[8], bf[8], bg[8], bo[8];
  #pragma unroll
  for (int i = 0; i < 8; ++i) {
    cc[i] = 0.0f;
    bi[i] = bias[0 * HP + n0 + i];
    bf[i] = bias[1 * HP + n0 + i];
    bg[i] = bias[2 * HP + n0 + i];
    bo[i] = bias[3 * HP + n0 + i];
  }
  {
    v8h z;
    #pragma unroll
    for (int i = 0; i < 8; ++i) z[i] = (_Float16)0.0f;
    *(v8h*)&hbuf[m * LDS_HSTRIDE + n0] = z;   // h0 = 0
  }
  __syncthreads();

  const size_t ioBase = ((size_t)(b0 + m) * Th) * HP + n0;

  for (int t = 0; t < Th; ++t) {
    // stage x_t tile (16x64 f16) into LDS; prefetch t+1 line
    *(v8h*)&xbuf[m * LDS_HSTRIDE + n0] = *(const v8h*)(in + ioBase + (size_t)t * HP);
    if (t + 1 < Th)
      __builtin_prefetch(in + ioBase + (size_t)(t + 1) * HP, 0, 1);
    __syncthreads();

    // gates(16x256) = [x_t | h_{t-1}] (16x128) @ Wcat (128x256), f32 accum
    v8f acc[4];
    #pragma unroll
    for (int j = 0; j < 4; ++j) {
      v8f z;
      #pragma unroll
      for (int v = 0; v < 8; ++v) z[v] = 0.0f;
      acc[j] = z;
    }
    #pragma unroll
    for (int kt = 0; kt < 4; ++kt) {
      const v16h a = (kt < 2) ? load_a_frag(xbuf, lane, kt * 32)
                              : load_a_frag(hbuf, lane, (kt - 2) * 32);
      #pragma unroll
      for (int j = 0; j < 4; ++j)
        acc[j] = __builtin_amdgcn_wmma_f32_16x16x32_f16(
            false, a, false, Bfrag[j][kt], (short)0, acc[j], false, false);
    }

    // scatter C/D fragments (VGPR v: lanes 0-15 -> M=v, lanes 16-31 -> M=v+8)
    {
      const int nc = lane & 15;
      const int mb = (lane >> 4) * 8;
      #pragma unroll
      for (int j = 0; j < 4; ++j) {
        const int ncol = (w * 4 + j) * 16 + nc;
        #pragma unroll
        for (int v = 0; v < 8; ++v)
          gbuf[(mb + v) * LDS_GSTRIDE + ncol] = acc[j][v];
      }
    }
    __syncthreads();

    // elementwise LSTM cell update; c lives in registers; branchless activations
    v8h hv;
    #pragma unroll
    for (int i = 0; i < 8; ++i) {
      const int n = n0 + i;
      const float ig = sigmoid_fast(gbuf[m * LDS_GSTRIDE + 0 * HP + n] + bi[i]);
      const float fg = sigmoid_fast(gbuf[m * LDS_GSTRIDE + 1 * HP + n] + bf[i]);
      const float gg = tanh_fast   (gbuf[m * LDS_GSTRIDE + 2 * HP + n] + bg[i]);
      const float og = sigmoid_fast(gbuf[m * LDS_GSTRIDE + 3 * HP + n] + bo[i]);
      const float c  = fg * cc[i] + ig * gg;
      cc[i] = c;
      hv[i] = (_Float16)(og * tanh_fast(c));
    }
    // h_t -> LDS (for t+1's WMMA; made visible by next iteration's barrier)
    // and -> global (input for layer l+1 / FC)
    *(v8h*)&hbuf[m * LDS_HSTRIDE + n0] = hv;
    *(v8h*)(out + ioBase + (size_t)t * HP) = hv;
  }
}

// Repack per-layer weights into concat padded f16 [KTOT][GP] + fused bias [GP].
// Column n = g*64 + j maps to original gate row r = g*49 + j (torch i,f,g,o order).
__global__ void prep_weights_kernel(const float* __restrict__ Wih0,
                                    const float* __restrict__ WihR,
                                    const float* __restrict__ Whh,
                                    const float* __restrict__ bih,
                                    const float* __restrict__ bhh,
                                    _Float16* __restrict__ Wws,
                                    float* __restrict__ Bws) {
  const int l   = blockIdx.y;
  const int idx = blockIdx.x * blockDim.x + threadIdx.x;  // 0..KTOT*GP-1
  const int k = idx >> 8;
  const int n = idx & 255;
  const int g = n >> 6;
  const int j = n & 63;
  float v = 0.0f;
  if (j < Hh) {
    const int r = g * Hh + j;
    if (k < HP) {                       // x part
      if (l == 0) { if (k < Ih) v = Wih0[r * Ih + k]; }
      else        { if (k < Hh) v = WihR[((size_t)(l - 1) * 4 * Hh + r) * Hh + k]; }
    } else {                            // h part
      const int kk = k - HP;
      if (kk < Hh) v = Whh[((size_t)l * 4 * Hh + r) * Hh + kk];
    }
  }
  Wws[(size_t)l * (KTOT * GP) + idx] = (_Float16)v;
  if (k == 0) {
    float bv = 0.0f;
    if (j < Hh) {
      const int r = g * Hh + j;
      bv = bih[l * 4 * Hh + r] + bhh[l * 4 * Hh + r];
    }
    Bws[l * GP + n] = bv;
  }
}

// fp32 [B,T,7] -> zero-padded f16 [B,T,64]
__global__ void pack_x_kernel(const float* __restrict__ x, _Float16* __restrict__ outp) {
  const size_t total = (size_t)Bh * Th * HP;
  for (size_t idx = (size_t)blockIdx.x * blockDim.x + threadIdx.x; idx < total;
       idx += (size_t)gridDim.x * blockDim.x) {
    const int k = (int)(idx & (HP - 1));
    const size_t bt = idx >> 6;
    float v = 0.0f;
    if (k < Ih) v = x[bt * Ih + k];
    outp[idx] = (_Float16)v;
  }
}

// out[b,t,o] = sum_k h[b,t,k] * fc_w[o,k] + fc_b[o]
__global__ void fc_kernel(const _Float16* __restrict__ h, const float* __restrict__ fcw,
                          const float* __restrict__ fcb, float* __restrict__ out) {
  const int idx = blockIdx.x * blockDim.x + threadIdx.x;
  if (idx >= Bh * Th * Oh) return;
  const int o = idx % Oh;
  const size_t bt = (size_t)idx / Oh;
  float s = fcb[o];
  #pragma unroll
  for (int k = 0; k < Hh; ++k)
    s += (float)h[bt * HP + k] * fcw[o * Hh + k];
  out[idx] = s;
}

extern "C" void kernel_launch(void* const* d_in, const int* in_sizes, int n_in,
                              void* d_out, int out_size, void* d_ws, size_t ws_size,
                              hipStream_t stream) {
  (void)in_sizes; (void)n_in; (void)out_size; (void)ws_size;
  const float* x    = (const float*)d_in[0];
  const float* Wih0 = (const float*)d_in[1];
  const float* WihR = (const float*)d_in[2];
  const float* Whh  = (const float*)d_in[3];
  const float* bih  = (const float*)d_in[4];
  const float* bhh  = (const float*)d_in[5];
  const float* fcw  = (const float*)d_in[6];
  const float* fcb  = (const float*)d_in[7];

  // ws layout: [weights f16 7*128*256][bias f32 7*256] ... @512KB: bufA, bufB (f16 [B,T,64])
  _Float16* Wws  = (_Float16*)d_ws;
  float*    Bws  = (float*)((char*)d_ws + (size_t)Lh * KTOT * GP * sizeof(_Float16));
  _Float16* bufA = (_Float16*)((char*)d_ws + (512u << 10));
  _Float16* bufB = bufA + (size_t)Bh * Th * HP;

  prep_weights_kernel<<<dim3(KTOT * GP / 256, Lh), 256, 0, stream>>>(
      Wih0, WihR, Whh, bih, bhh, Wws, Bws);
  pack_x_kernel<<<8192, 256, 0, stream>>>(x, bufA);

  const _Float16* cur = bufA;
  _Float16*       nxt = bufB;
  for (int l = 0; l < Lh; ++l) {
    lstm_layer_kernel<<<Bh / 16, 128, 0, stream>>>(
        cur, nxt, Wws + (size_t)l * KTOT * GP, Bws + (size_t)l * GP);
    const _Float16* tmp = nxt;
    nxt = (_Float16*)cur;
    cur = tmp;
  }
  fc_kernel<<<(Bh * Th * Oh + 255) / 256, 256, 0, stream>>>(cur, fcw, fcb, (float*)d_out);
}
